// QuaternionLinearAutograd_62637803045343
// MI455X (gfx1250) — matrix operand order repaired and checked by reading, not compiled, not run on hardware
//
#include <hip/hip_runtime.h>
#include <hip/hip_bf16.h>

// CDNA5 / gfx1250 quaternion linear layer via bf16x3 fp32-emulated WMMA GEMM.
// C(8192x12288) = X(8192x4096) @ HamiltonBlockGather(r,i,j,k)(4096x12288)
// K loop split into 4 quadrant segments with COMPILE-TIME weight-pointer
// selection (s' = qi^qo iterated as a constant); software-pipelined
// global->reg->LDS, double-buffered LDS, one barrier per K-chunk.
// sched_barrier(0) pins next-chunk conversion VALU after current-chunk WMMAs.

typedef __attribute__((ext_vector_type(16))) __bf16 v16bf;
typedef __attribute__((ext_vector_type(8)))  __bf16 v8bf;
typedef __attribute__((ext_vector_type(4)))  __bf16 v4bf;
typedef __attribute__((ext_vector_type(8)))  float  v8f;

union V16 { v16bf v; v8bf h[2]; };

#define BM     64      // block tile M
#define BN     128     // block tile N
#define BK     32      // K chunk (one bf16 WMMA K)
#define LDT    40      // padded LDS row stride in bf16 elements (80 B, bank-spread)
#define KDIM   4096    // 4 * IN_FEATURES
#define SEGC   32      // chunks per quadrant segment (1024 / BK)
#define NW     3072    // weight tensor row stride (OUT_FEATURES)
#define OUTC   4096    // columns per output slice

__global__ __launch_bounds__(256)
void quat_linear_wmma(const float* __restrict__ x,
                      const float* __restrict__ wr,
                      const float* __restrict__ wi,
                      const float* __restrict__ wj,
                      const float* __restrict__ wk,
                      float* __restrict__ out)
{
    // double-buffered LDS: 2 * (A 10240 B + B 20480 B) = 61440 B
    __shared__ __bf16 Ahi[2][BM * LDT], Alo[2][BM * LDT];
    __shared__ __bf16 Bhi[2][BN * LDT], Blo[2][BN * LDT];

    const int t    = threadIdx.x;
    const int lane = t & 31;
    const int wid  = t >> 5;
    const int wm   = wid >> 2;          // wave M position: 0..1
    const int wn   = wid & 3;           // wave N position: 0..3
    const int m0   = blockIdx.y * BM;
    const int n0   = blockIdx.x * BN;   // block never straddles 1024/4096 boundaries

    // ---- scalar (blockIdx-only) quaternion column decomposition ----
    const int qo   = (n0 & 4095) >> 10;                 // output quadrant (scalar)
    const int scol = (n0 >> 12) * 1024 + (n0 & 1023);   // scalar part of weight column

    // ---- loop-invariant per-lane offsets ----
    const int r0   = t >> 3;            // A staging row, jj=0
    const int k4_  = (t & 7) * 4;
    const int aoff0 = r0 * KDIM + k4_;
    const int aoff1 = (r0 + 32) * KDIM + k4_;
    const int aldsOff0 = r0 * LDT + k4_;
    const int aldsOff1 = (r0 + 32) * LDT + k4_;
    const int bn   = t & 127;           // B staging column
    const int kb   = (t >> 7) * 16;     // B staging K span start {0,16}
    const int bvoff = bn + kb * NW;     // per-lane element offset vs scalar B base
    const int bldsOff = bn * LDT + kb;

    v8f acc[2][2];
    #pragma unroll
    for (int a = 0; a < 2; ++a)
        #pragma unroll
        for (int b = 0; b < 2; ++b)
            acc[a][b] = (v8f){0.f,0.f,0.f,0.f,0.f,0.f,0.f,0.f};

    // ---------- pipeline stage helpers ----------
    auto loadA = [&](float4 (&xa)[2], int kcL) {
        const float* abase = x + (size_t)m0 * KDIM + kcL * BK;   // scalar base
        __builtin_prefetch(abase + aoff0 + BK, 0, 1);            // next chunk
        xa[0] = *(const float4*)(abase + aoff0);
        xa[1] = *(const float4*)(abase + aoff1);
    };

    auto loadB = [&](float (&wv)[16], const float* wsel, int ic) {
        // segment start is 1024-aligned -> weight row base = ic*BK (pure induction)
        const float* sbase = wsel + (size_t)(ic * BK) * NW + scol;
        #pragma unroll
        for (int k = 0; k < 16; ++k)
            wv[k] = sbase[bvoff + k * NW];          // imm offset = k*NW*4 bytes
    };

    auto stage = [&](const float4 (&xa)[2], const float (&wv)[16], float sg, int buf) {
        #pragma unroll
        for (int jj = 0; jj < 2; ++jj) {
            float fx = xa[jj].x, fy = xa[jj].y, fz = xa[jj].z, fw = xa[jj].w;
            v4bf h4, l4;
            { __bf16 h = (__bf16)fx; h4[0] = h; l4[0] = (__bf16)(fx - (float)h); }
            { __bf16 h = (__bf16)fy; h4[1] = h; l4[1] = (__bf16)(fy - (float)h); }
            { __bf16 h = (__bf16)fz; h4[2] = h; l4[2] = (__bf16)(fz - (float)h); }
            { __bf16 h = (__bf16)fw; h4[3] = h; l4[3] = (__bf16)(fw - (float)h); }
            int o = jj ? aldsOff1 : aldsOff0;
            *(v4bf*)&Ahi[buf][o] = h4;
            *(v4bf*)&Alo[buf][o] = l4;
        }
        v8bf bh0, bh1, bl0, bl1;
        #pragma unroll
        for (int k = 0; k < 16; ++k) {
            float w = sg * wv[k];
            __bf16 h = (__bf16)w;
            __bf16 l = (__bf16)(w - (float)h);
            if (k < 8) { bh0[k] = h;     bl0[k] = l; }
            else       { bh1[k - 8] = h; bl1[k - 8] = l; }
        }
        *(v8bf*)&Bhi[buf][bldsOff]     = bh0;
        *(v8bf*)&Bhi[buf][bldsOff + 8] = bh1;
        *(v8bf*)&Blo[buf][bldsOff]     = bl0;
        *(v8bf*)&Blo[buf][bldsOff + 8] = bl1;
    };

    // A k order used here (matches B): lanes 0-15 hold K 0..15, lanes 16-31 K 16..31
    const int ka  = (lane >> 4) * 16;
    // B k order used here (matches A): lanes 0-15 hold K 0..15, lanes 16-31 K 16..31
    const int kbb = (lane >> 4) * 16;

    auto compute = [&](int buf) {
        V16 ah[2], al[2], bh[2], bl[2];
        #pragma unroll
        for (int tm = 0; tm < 2; ++tm) {
            int row = (wm * 32 + tm * 16 + (lane & 15)) * LDT;
            ah[tm].h[0] = *(const v8bf*)&Ahi[buf][row + ka];
            ah[tm].h[1] = *(const v8bf*)&Ahi[buf][row + ka + 8];
            al[tm].h[0] = *(const v8bf*)&Alo[buf][row + ka];
            al[tm].h[1] = *(const v8bf*)&Alo[buf][row + ka + 8];
        }
        #pragma unroll
        for (int tn = 0; tn < 2; ++tn) {
            int rowb = (wn * 32 + tn * 16 + (lane & 15)) * LDT;
            bh[tn].h[0] = *(const v8bf*)&Bhi[buf][rowb + kbb];
            bh[tn].h[1] = *(const v8bf*)&Bhi[buf][rowb + kbb + 8];
            bl[tn].h[0] = *(const v8bf*)&Blo[buf][rowb + kbb];
            bl[tn].h[1] = *(const v8bf*)&Blo[buf][rowb + kbb + 8];
        }
        #pragma unroll
        for (int tm = 0; tm < 2; ++tm)
            #pragma unroll
            for (int tn = 0; tn < 2; ++tn) {
                acc[tm][tn] = __builtin_amdgcn_wmma_f32_16x16x32_bf16(
                    false, ah[tm].v, false, bl[tn].v, (short)0, acc[tm][tn], false, false);
                acc[tm][tn] = __builtin_amdgcn_wmma_f32_16x16x32_bf16(
                    false, al[tm].v, false, bh[tn].v, (short)0, acc[tm][tn], false, false);
                acc[tm][tn] = __builtin_amdgcn_wmma_f32_16x16x32_bf16(
                    false, ah[tm].v, false, bh[tn].v, (short)0, acc[tm][tn], false, false);
            }
    };

    // One quadrant segment: 32 chunks, fixed weight tensor + sign, pipelined x2.
    auto runSegment = [&](const float* wsel, int qi, float sgn) {
        const int kc0 = qi * SEGC;          // scalar chunk offset of this segment
        float4 xa0[2], xa1[2];
        float  wv0[16], wv1[16];
        loadA(xa0, kc0);
        loadB(wv0, wsel, 0);
        for (int ic = 0; ic < SEGC; ic += 2) {
            stage(xa0, wv0, sgn, 0);
            __syncthreads();
            loadA(xa1, kc0 + ic + 1);
            loadB(wv1, wsel, ic + 1);
            compute(0);
            __builtin_amdgcn_sched_barrier(0);   // keep slot1 conversions after these WMMAs

            stage(xa1, wv1, sgn, 1);
            __syncthreads();
            int in2 = (ic + 2 < SEGC) ? ic + 2 : 0;   // clamp: harmless re-load
            loadA(xa0, kc0 + in2);
            loadB(wv0, wsel, in2);
            compute(1);
            __builtin_amdgcn_sched_barrier(0);   // keep slot0 conversions after these WMMAs
        }
    };

    // ---------- segments: s' = qi^qo iterated as a COMPILE-TIME constant ----------
    #pragma unroll
    for (int sp = 0; sp < 4; ++sp) {
        const float* wsel = (sp == 0) ? wr : (sp == 1) ? wi : (sp == 2) ? wj : wk;  // folds
        int   qi  = sp ^ qo;                                    // scalar
        // sign: -1 at (qi,qo) in {(1,3),(2,1),(3,2)} -> bit positions 7, 9, 14
        float sgn = ((0x4280u >> (qi * 4 + qo)) & 1u) ? -1.f : 1.f;  // scalar, per segment
        runSegment(wsel, qi, sgn);
    }

    // ---- epilogue: C/D layout — VGPR j: lanes 0-15 M=j, lanes 16-31 M=j+8; N = lane&15 ----
    float* obase = out + ((size_t)(n0 >> 12) << 25) + (n0 & 4095);
    #pragma unroll
    for (int tm = 0; tm < 2; ++tm)
        #pragma unroll
        for (int tn = 0; tn < 2; ++tn)
            #pragma unroll
            for (int j = 0; j < 8; ++j) {
                int m    = m0 + wm * 32 + tm * 16 + j + ((lane >> 4) << 3);
                int ncol = wn * 32 + tn * 16 + (lane & 15);
                obase[(size_t)m * OUTC + ncol] = acc[tm][tn][j];
            }
}

extern "C" void kernel_launch(void* const* d_in, const int* in_sizes, int n_in,
                              void* d_out, int out_size, void* d_ws, size_t ws_size,
                              hipStream_t stream) {
    (void)in_sizes; (void)n_in; (void)out_size; (void)d_ws; (void)ws_size;
    const float* x  = (const float*)d_in[0];
    const float* wr = (const float*)d_in[1];
    const float* wi = (const float*)d_in[2];
    const float* wj = (const float*)d_in[3];
    const float* wk = (const float*)d_in[4];
    float* out = (float*)d_out;
    dim3 grid(12288 / BN, 8192 / BM);   // 96 x 128 blocks
    quat_linear_wmma<<<grid, dim3(256), 0, stream>>>(x, wr, wi, wj, wk, out);
}
